// LowRankGDN_68942815035715
// MI455X (gfx1250) — matrix operand order repaired and checked by reading, not compiled, not run on hardware
//
#include <hip/hip_runtime.h>

typedef __attribute__((ext_vector_type(16))) _Float16 v16h;
typedef __attribute__((ext_vector_type(2)))  _Float16 v2h;
typedef __attribute__((ext_vector_type(8)))  float    v8f;

#define PEDESTAL   1.4551915228366852e-11f   // 2^-36
#define BOUND_BETA 1.0000072759321560e-03f   // sqrt(1e-6 + 2^-36)
#define BOUND_A    3.8146972656250000e-06f   // sqrt(2^-36) = 2^-18

#define C_DIM        192
#define R_DIM        8
#define N_TILES      16384        // 8 * 65536 / 32
#define TILES_PER_WG 16
#define N_WG         (N_TILES / TILES_PER_WG)
#define M_HALVES     (C_DIM * C_DIM)   // 36864

__device__ __forceinline__ float nn_reparam(float p, float bound) {
    float m = fmaxf(p, bound);
    return m * m - PEDESTAL;
}

// ---------------------------------------------------------------------------
// Setup: M = A_r * A_r^T (f16, pre-swizzled into WMMA A-operand layout) + beta_r
// A-operand half-index = ((cb*6 + chunk)*32 + lane)*16 + e
//   c  = cb*16 + (lane & 15)
//   k  = (e < 8 ? e : e + 8) + ((lane >> 4) << 3)   // ISA 16-bit A 16x32 layout
//   c2 = chunk*32 + k
// ---------------------------------------------------------------------------
__global__ void lrgdn_setup(const float* __restrict__ A,
                            const float* __restrict__ beta,
                            _Float16* __restrict__ Mswz,
                            float* __restrict__ beta_r) {
    int tid = blockIdx.x * 256 + threadIdx.x;
    if (tid < M_HALVES) {
        int e     = tid & 15;
        int lane  = (tid >> 4) & 31;
        int ci    = tid >> 9;          // 0..71
        int chunk = ci % 6;
        int cb    = ci / 6;
        int c  = cb * 16 + (lane & 15);
        int k  = (e < 8 ? e : e + 8) + ((lane >> 4) << 3);
        int c2 = chunk * 32 + k;
        float acc = 0.0f;
#pragma unroll
        for (int r = 0; r < R_DIM; ++r) {
            float a1 = nn_reparam(A[c  * R_DIM + r], BOUND_A);
            float a2 = nn_reparam(A[c2 * R_DIM + r], BOUND_A);
            acc = fmaf(a1, a2, acc);
        }
        Mswz[tid] = (_Float16)acc;
    } else if (tid < M_HALVES + C_DIM) {
        int c = tid - M_HALVES;
        beta_r[c] = nn_reparam(beta[c], BOUND_BETA);
    }
}

// ---------------------------------------------------------------------------
// Main: per WG (12 waves), wave w owns output channel block [16w, 16w+16).
// denom = M*x^2 + beta via v_wmma_f32_16x16x32_f16, out = x * rsqrt(denom).
// ---------------------------------------------------------------------------
__global__ __launch_bounds__(384)
void lrgdn_main(const float* __restrict__ x,
                const _Float16* __restrict__ Mswz,
                const float* __restrict__ beta_r,
                float* __restrict__ out) {
    // x^2 staged in exact B-operand layout: [(sub*6+chunk)*32 + lane][16 halves]
    __shared__ __align__(32) _Float16 x2s[12 * 32 * 16];

    const int t    = threadIdx.x;
    const int lane = t & 31;
    const int w    = t >> 5;                 // wave id == channel block 0..11
    const int hi8  = (lane >> 4) << 3;       // 8 for upper half-wave

    // Per-wave constant A operands (full M row-block, 6 K-chunks, 48 VGPRs)
    const v16h* Mv = (const v16h*)Mswz;
    v16h aop[6];
#pragma unroll
    for (int ck = 0; ck < 6; ++ck)
        aop[ck] = Mv[(w * 6 + ck) * 32 + lane];

    // beta broadcast matching C/D layout: VGPR v -> c = 16w + v + hi8
    float bv[8];
#pragma unroll
    for (int v = 0; v < 8; ++v)
        bv[v] = beta_r[w * 16 + v + hi8];

    const v16h* x2v = (const v16h*)x2s;

    for (int it = 0; it < TILES_PER_WG; ++it) {
        const int idx  = blockIdx.x * TILES_PER_WG + it;
        const int nimg = idx >> 11;                // 2048 tiles per image plane
        const int hw0  = (idx & 2047) << 5;        // 32 pixels per tile
        const size_t xbase = ((size_t)nimg * C_DIM) << 16;

        __syncthreads();   // previous iteration's LDS readers done

        // ---- stage x^2 (f16, packed channel pairs) into LDS in B-operand layout
        {
            const int p    = t & 31;
            const int cp   = t >> 5;       // channel-pair group 0..11
            const int sub  = p >> 4;
            const int ncol = p & 15;
            const bool pf  = (idx + 1) < N_TILES;
            const int idx2 = idx + 1;
            const size_t xbase2 = ((size_t)(idx2 >> 11) * C_DIM) << 16;
            const int hw02 = (idx2 & 2047) << 5;
#pragma unroll
            for (int i = 0; i < 8; ++i) {
                int c2 = 2 * cp + 24 * i;            // even channel of the pair
                size_t g0 = xbase + ((size_t)c2 << 16) + (size_t)(hw0 + p);
                float xv0 = x[g0];
                float xv1 = x[g0 + 65536];           // c2 + 1
                if (pf) {  // prefetch next tile -> global_prefetch_b8
                    __builtin_prefetch(&x[xbase2 + ((size_t)c2 << 16) + (size_t)(hw02 + p)], 0, 1);
                    __builtin_prefetch(&x[xbase2 + ((size_t)(c2 + 1) << 16) + (size_t)(hw02 + p)], 0, 1);
                }
                int chunk = c2 >> 5;
                int k     = c2 & 31;                 // even -> k and k+1 adjacent halves
                int hidx  = (((sub * 6 + chunk) * 32) + ncol + ((k & 8) << 1)) * 16 + (k & 7) + ((k & 16) >> 1);
                v2h pr;
                pr.x = (_Float16)(xv0 * xv0);
                pr.y = (_Float16)(xv1 * xv1);
                *(v2h*)&x2s[hidx] = pr;              // single ds_store_b32
            }
        }
        __syncthreads();

        // ---- GEMM + epilogue: two 16-pixel sub-tiles per wave
#pragma unroll
        for (int sub = 0; sub < 2; ++sub) {
            v8f acc;
#pragma unroll
            for (int v = 0; v < 8; ++v) acc[v] = bv[v];

#pragma unroll
            for (int ck = 0; ck < 6; ++ck) {
                v16h b = x2v[(sub * 6 + ck) * 32 + lane];
                acc = __builtin_amdgcn_wmma_f32_16x16x32_f16(
                    false, aop[ck], false, b, (short)0, acc, false, false);
            }

            const int pix = hw0 + sub * 16 + (lane & 15);
#pragma unroll
            for (int v = 0; v < 8; ++v) {
                int c = w * 16 + v + hi8;
                size_t o = xbase + ((size_t)c << 16) + (size_t)pix;
                // denom >= beta_r ~ 1.0 here: raw v_rsq_f32, no denorm guard needed
                out[o] = x[o] * __builtin_amdgcn_rsqf(acc[v]);
            }
        }
    }
}

extern "C" void kernel_launch(void* const* d_in, const int* in_sizes, int n_in,
                              void* d_out, int out_size, void* d_ws, size_t ws_size,
                              hipStream_t stream) {
    const float* x    = (const float*)d_in[0];
    const float* beta = (const float*)d_in[1];
    const float* A    = (const float*)d_in[2];
    float* outp       = (float*)d_out;

    _Float16* Mswz  = (_Float16*)d_ws;                         // 73728 B
    float*    betar = (float*)((char*)d_ws + M_HALVES * 2);    // 768 B

    lrgdn_setup<<<(M_HALVES + C_DIM + 255) / 256, 256, 0, stream>>>(A, beta, Mswz, betar);
    lrgdn_main<<<N_WG, 384, 0, stream>>>(x, Mswz, betar, outp);
}
